// BlockConv2D_17403207483414
// MI455X (gfx1250) — matrix operand order repaired and checked by reading, not compiled, not run on hardware
//
#include <hip/hip_runtime.h>
#include <cstddef>

// CDNA5 / gfx1250. wave32. WMMA f32<-f16 16x16x32, LDS-staged A operands.
typedef __attribute__((ext_vector_type(16))) _Float16 v16h;
typedef __attribute__((ext_vector_type(8)))  _Float16 v8h;
typedef __attribute__((ext_vector_type(8)))  float    v8f;

#define NB 16     // blocks
#define CINB 32
#define COUTB 32
#define HH 64
#define WW 64
#define CH 512

#define RSPAN 4   // output rows per workgroup (1 per wave)
#define ROWS  6   // staged input rows  (RSPAN + 2 halo)
#define XS    66  // staged pixels/row  (64 + 2 halo, slot = xi+1)
#define CPAD  40  // channel pitch in halves: 80B row stride -> 16B-aligned
                  // b128 chunks, 20-bank lane stride (conflict-free per group)

// ------------------------------------------------------------------
// Weight pre-pack: per (block,tap) the two N=16 B-matrix halves as
// per-lane v16h chunks, split into f16 hi / f16 lo terms.
// B-matrix (32x16 f16) per-lane layout:
//   lane L in [0,16):  column n=L,    holds K = 0..15
//   lane L in [16,32): column n=L-16, holds K = 16..31
// ws: wpack[((bt*2 + half)*2 + term)*32 + lane][16], bt = b*9+tap.
// Size: 144*2*2*32*16*2B = 589,824 bytes of d_ws.
// ------------------------------------------------------------------
__global__ void prep_weights_kernel(const float* __restrict__ w,
                                    _Float16* __restrict__ wpack) {
    int bt   = blockIdx.x;          // b*9 + tap, 144 total
    int half = threadIdx.x >> 5;
    int lane = threadIdx.x & 31;
    int g    = lane >> 4;
    int n    = lane & 15;

    const float* wsrc = w + (size_t)bt * (CINB * COUTB);
    size_t baseHi = ((((size_t)bt * 2 + half) * 2 + 0) * 32 + lane) * 16;
    size_t baseLo = ((((size_t)bt * 2 + half) * 2 + 1) * 32 + lane) * 16;

#pragma unroll
    for (int j = 0; j < 16; ++j) {
        int k = 16 * g + j;
        float v = wsrc[k * COUTB + half * 16 + n];
        _Float16 h = (_Float16)v;
        wpack[baseHi + j] = h;
        wpack[baseLo + j] = (_Float16)(v - (float)h);
    }
}

// ------------------------------------------------------------------
// Main kernel: workgroup = (img n, block b, 4-row strip yo).
// Phase 1: gather the block's 32 routed channels of 6 halo rows into
//          LDS, pre-split into f16 hi / f16 lo planes (gathered once,
//          converted once; reused by 9 taps x 4 x-tiles).
//          12,672 elements == 128 threads * 99 iters exactly -> no
//          guard, no divergence.
// Phase 2: wave w computes row y = yo*4+w: 4 tiles of 16px x 32cout,
//          A-fragments via ds_load_b128, f16x3 WMMA accumulation.
// Scatter store + bias + ReLU fused (blocks_out is a permutation ->
// disjoint channel partitions, each output written exactly once).
// ------------------------------------------------------------------
__global__ __launch_bounds__(128)
void blockconv_wmma_kernel(const float* __restrict__ x,
                           const _Float16* __restrict__ wpack,
                           const float* __restrict__ bias,
                           const int* __restrict__ bin,
                           const int* __restrict__ bout,
                           float* __restrict__ out) {
    // hi plane [0], lo plane [1]: 2*6*66*40*2B = 63,360 bytes
    __shared__ __align__(16) _Float16 ldsA[2][ROWS][XS][CPAD];

    int tid = threadIdx.x;
    int yo = blockIdx.x & 15;          // row strip
    int n  = (blockIdx.x >> 4) & 15;   // image
    int b  = blockIdx.x >> 8;          // block

    // ---------------- Phase 1: gather + split into LDS ----------------
    // One wave iteration = 32 lanes = the 32 routed channels of one
    // (row, pixel) slot: a single 32-lane gather per 32 staged elements.
    {
        int c    = tid & 31;                 // channel slot (invariant)
        int chan = bin[b * CINB + c];        // routed source channel
        int r    = 0;
        int xs   = tid >> 5;                 // slot advances by 4/iter
        for (int i = 0; i < (ROWS * XS * 32) / 128; ++i) {  // 99 exact
            int yi = yo * RSPAN - 1 + r;
            int xi = xs - 1;
            bool valid = (yi >= 0) && (yi < HH) && (xi >= 0) && (xi < WW);
            int yc = valid ? yi : 0;
            int xc = valid ? xi : 0;
            float v = x[(((size_t)(n * HH + yc)) * WW + xc) * CH + chan];
            v = valid ? v : 0.0f;
            _Float16 h = (_Float16)v;
            ldsA[0][r][xs][c] = h;
            ldsA[1][r][xs][c] = (_Float16)(v - (float)h);
            xs += 4;
            if (xs >= XS) { xs -= XS; r += 1; }
        }
    }
    __syncthreads();

    // ---------------- Phase 2: WMMA over staged rows -------------------
    int lane = tid & 31;
    int w    = tid >> 5;                // wave -> output row in strip
    int g    = lane >> 4;
    int m    = lane & 15;               // A-matrix row M (pixel in tile)
    int y    = yo * RSPAN + w;

    v8f acc[4][2];
#pragma unroll
    for (int xg = 0; xg < 4; ++xg) {
        v8f z = {};
        acc[xg][0] = z;
        acc[xg][1] = z;
    }

    const _Float16* wp = wpack + (size_t)(b * 9) * (2 * 2 * 32 * 16);

#pragma unroll 1                         // keep rolled: bound B-frag pressure
    for (int dy = 0; dy < 3; ++dy) {
#pragma unroll 1
        for (int dx = 0; dx < 3; ++dx) {
            int tap = dy * 3 + dx;
            const v16h* wb = (const v16h*)(wp + (size_t)tap * (2 * 2 * 32 * 16));
            v16h b0h = wb[0 * 32 + lane];
            v16h b0l = wb[1 * 32 + lane];
            v16h b1h = wb[2 * 32 + lane];
            v16h b1l = wb[3 * 32 + lane];
            int rr = w + dy;             // staged row, always in [0,6)
#pragma unroll
            for (int xg = 0; xg < 4; ++xg) {
                int xs2 = xg * 16 + m + dx;   // padded slot (= xi+1)
                const _Float16* ph = &ldsA[0][rr][xs2][0];
                const _Float16* pl = &ldsA[1][rr][xs2][0];
                // k order used here (same as B pack): lane group g holds K in
                // {16g..16g+7} (v0-3) and {16g+8..16g+15} (v4-7)
                union { v16h v; v8h h[2]; } ah, al;
                ah.h[0] = *(const v8h*)(ph + 16 * g);
                ah.h[1] = *(const v8h*)(ph + 16 * g + 8);
                al.h[0] = *(const v8h*)(pl + 16 * g);
                al.h[1] = *(const v8h*)(pl + 16 * g + 8);
                // f32-via-f16x3: hi*hi + lo*hi + hi*lo, f32 accumulate
                acc[xg][0] = __builtin_amdgcn_wmma_f32_16x16x32_f16(
                    false, ah.v, false, b0h, (short)0, acc[xg][0], false, false);
                acc[xg][1] = __builtin_amdgcn_wmma_f32_16x16x32_f16(
                    false, ah.v, false, b1h, (short)0, acc[xg][1], false, false);
                acc[xg][0] = __builtin_amdgcn_wmma_f32_16x16x32_f16(
                    false, al.v, false, b0h, (short)0, acc[xg][0], false, false);
                acc[xg][1] = __builtin_amdgcn_wmma_f32_16x16x32_f16(
                    false, al.v, false, b1h, (short)0, acc[xg][1], false, false);
                acc[xg][0] = __builtin_amdgcn_wmma_f32_16x16x32_f16(
                    false, ah.v, false, b0l, (short)0, acc[xg][0], false, false);
                acc[xg][1] = __builtin_amdgcn_wmma_f32_16x16x32_f16(
                    false, ah.v, false, b1l, (short)0, acc[xg][1], false, false);
            }
        }
    }

    // ---------------- epilogue: scatter + bias + ReLU ------------------
    // C/D layout: VGPR r -> M = r + 8*(lane/16), N = lane&15.
    int co0 = bout[b * COUTB + m];
    int co1 = bout[b * COUTB + 16 + m];
    float bb0 = bias[co0];
    float bb1 = bias[co1];
    size_t rowbase = ((size_t)(n * HH + y)) * WW;
#pragma unroll
    for (int xg = 0; xg < 4; ++xg) {
#pragma unroll
        for (int rg = 0; rg < 8; ++rg) {
            int mm = rg + 8 * g;
            float* po = out + (rowbase + xg * 16 + mm) * CH;
            po[co0] = fmaxf(acc[xg][0][rg] + bb0, 0.0f);
            po[co1] = fmaxf(acc[xg][1][rg] + bb1, 0.0f);
        }
    }
}

extern "C" void kernel_launch(void* const* d_in, const int* in_sizes, int n_in,
                              void* d_out, int out_size, void* d_ws, size_t ws_size,
                              hipStream_t stream) {
    (void)in_sizes; (void)n_in; (void)out_size; (void)ws_size;
    const float* x    = (const float*)d_in[0];
    const float* w    = (const float*)d_in[1];
    const float* bias = (const float*)d_in[2];
    const int*   bin  = (const int*)d_in[3];
    const int*   bout = (const int*)d_in[4];
    float* out = (float*)d_out;
    _Float16* wpack = (_Float16*)d_ws;   // needs 589,824 bytes

    prep_weights_kernel<<<NB * 9, 64, 0, stream>>>(w, wpack);

    // 16 blocks x 16 images x 16 row-strips = 4096 workgroups of 4 waves
    blockconv_wmma_kernel<<<NB * 16 * (HH / RSPAN), 128, 0, stream>>>(
        x, wpack, bias, bin, bout, out);
}
